// dMaSIFConv_59768764891786
// MI455X (gfx1250) — compile-verified, matrix operand order checked by reading, not run on hardware
//
#include <hip/hip_runtime.h>
#include <hip/hip_bf16.h>
#include <math.h>

typedef __attribute__((ext_vector_type(16))) _Float16 v16h;
typedef __attribute__((ext_vector_type(8)))  float    v8f;

#define N_PTS 4096
#define I_DIM 16
#define H_DIM 64
#define O_DIM 16
#define C_DIM 8

__device__ __forceinline__ float leaky(float x) { return x > 0.0f ? x : 0.2f * x; }

// ---------------- stats zero ----------------
__global__ void k_zero_stats(float* stats) {
    if (threadIdx.x < 16) stats[threadIdx.x] = 0.0f;
}

// ---------------- net_in MLP: (N,16)->(N,64), accumulate group stats ----------------
__global__ __launch_bounds__(256) void k_net_in(const float* __restrict__ feat,
                                                const float* __restrict__ W1, const float* __restrict__ b1,
                                                const float* __restrict__ W2, const float* __restrict__ b2,
                                                float* __restrict__ fpre, float* __restrict__ stats) {
    int n = blockIdx.x * blockDim.x + threadIdx.x;
    if (n >= N_PTS) return;
    float x[I_DIM];
#pragma unroll
    for (int k = 0; k < I_DIM; ++k) x[k] = feat[n * I_DIM + k];
    float h1[H_DIM];
    for (int o = 0; o < H_DIM; ++o) {
        float a = b1[o];
#pragma unroll
        for (int k = 0; k < I_DIM; ++k) a += W1[o * I_DIM + k] * x[k];
        h1[o] = leaky(a);
    }
    float gs[4] = {0.f, 0.f, 0.f, 0.f}, gss[4] = {0.f, 0.f, 0.f, 0.f};
    for (int o = 0; o < H_DIM; ++o) {
        float a = b2[o];
        for (int m = 0; m < H_DIM; ++m) a += W2[o * H_DIM + m] * h1[m];
        a = leaky(a);
        fpre[n * H_DIM + o] = a;
        int g = o >> 4;
        gs[g] += a; gss[g] += a * a;
    }
#pragma unroll
    for (int g = 0; g < 4; ++g) {
        atomicAdd(&stats[2 * g + 0], gs[g]);
        atomicAdd(&stats[2 * g + 1], gss[g]);
    }
}

// ---------------- GroupNorm apply; optionally write transposed (C x N) ----------------
__global__ __launch_bounds__(256) void k_gn_apply(const float* __restrict__ buf,
                                                  const float* __restrict__ stats,
                                                  const float* __restrict__ gamma, const float* __restrict__ beta,
                                                  float* __restrict__ outbuf, int C, int Cg, int transpose) {
    int idx = blockIdx.x * blockDim.x + threadIdx.x;
    if (idx >= N_PTS * C) return;
    int n = idx / C;
    int c = idx % C;
    int g = c / Cg;
    float M = (float)(Cg * N_PTS);
    float mu = stats[2 * g + 0] / M;
    float var = stats[2 * g + 1] / M - mu * mu;
    float x = buf[idx];
    float xn = (x - mu) * rsqrtf(var + 1e-5f);
    float y = xn * gamma[c] + beta[c];
    if (transpose) outbuf[c * N_PTS + n] = y;
    else           outbuf[idx] = y;
}

// ---------------- hot pairwise kernel: one wave per output point i ----------------
// res[i,h] = sum_j exp(-|dij|^2 (2 - ni.nj)^2) * relu(A2 relu(A1 (nuv_i dij) + B1) + B2)[h] * f[j,h]
// 8 -> 64 expansion on v_wmma_f32_16x16x32_f16 (K padded 8->32); B2 folded into the
// WMMA C accumulator (row-invariant broadcast in C/D layout), so D = A2.X8 + B2 directly.
__global__ __launch_bounds__(256) void k_pair_conv(const float* __restrict__ pts,
                                                   const float* __restrict__ nuv,
                                                   const float* __restrict__ A1, const float* __restrict__ B1,
                                                   const float* __restrict__ A2, const float* __restrict__ B2,
                                                   const float* __restrict__ fT,   // transposed (H, N)
                                                   float* __restrict__ res) {
    const int lane = threadIdx.x & 31;
    const int wave = threadIdx.x >> 5;
    const int i    = blockIdx.x * 8 + wave;
    const int l15  = lane & 15;
    const int half = lane >> 4;
    const float scale = 0.70710678118654752440f; // 1/(sqrt(2)*RADIUS)

    // per-i constants (uniform across wave)
    const float pix = pts[i * 3 + 0] * scale;
    const float piy = pts[i * 3 + 1] * scale;
    const float piz = pts[i * 3 + 2] * scale;
    float R[9];
#pragma unroll
    for (int r = 0; r < 9; ++r) R[r] = nuv[i * 9 + r];
    const float nix = R[0], niy = R[1], niz = R[2]; // normals = nuv[:,0,:]

    float a1[24], b1c[8];
#pragma unroll
    for (int r = 0; r < 24; ++r) a1[r] = A1[r];
#pragma unroll
    for (int c = 0; c < 8; ++c) b1c[c] = B1[c];

    // B fragments: B[k=c][n=h] = A2[h, c], K padded 8 -> 32 with zeros
    v16h bfrag[4];
#pragma unroll
    for (int ht = 0; ht < 4; ++ht) {
        v16h b = {};
        if (lane < 16) {
#pragma unroll
            for (int e = 0; e < 8; ++e)
                b[e] = (_Float16)A2[(ht * 16 + l15) * C_DIM + e];
        }
        bfrag[ht] = b;
    }

    // C fragments: broadcast B2 along M (element value depends only on N = lane&15)
    v8f cfrag[4];
#pragma unroll
    for (int ht = 0; ht < 4; ++ht) {
        const float b2 = B2[ht * 16 + l15];
        v8f c;
#pragma unroll
        for (int e = 0; e < 8; ++e) c[e] = b2;
        cfrag[ht] = c;
    }

    float racc[4] = {0.f, 0.f, 0.f, 0.f};

    for (int jb = 0; jb < N_PTS; jb += 32) {
        const int j = jb + lane; // all 32 lanes handle distinct j
        const float dx = pts[j * 3 + 0] * scale - pix;
        const float dy = pts[j * 3 + 1] * scale - piy;
        const float dz = pts[j * 3 + 2] * scale - piz;
        const float sq = dx * dx + dy * dy + dz * dz;
        const float dot = nix * nuv[j * 9 + 0] + niy * nuv[j * 9 + 1] + niz * nuv[j * 9 + 2];
        const float gg = 2.0f - dot;
        const float window = __expf(-sq * gg * gg);

        // local-basis coords
        const float X0 = R[0] * dx + R[1] * dy + R[2] * dz;
        const float X1 = R[3] * dx + R[4] * dy + R[5] * dz;
        const float X2 = R[6] * dx + R[7] * dy + R[8] * dz;

        // 3 -> 8 relu transform (per pair)
        float x8[8];
#pragma unroll
        for (int c = 0; c < 8; ++c) {
            float v = a1[c * 3 + 0] * X0 + a1[c * 3 + 1] * X1 + a1[c * 3 + 2] * X2 + b1c[c];
            x8[c] = fmaxf(v, 0.0f);
        }

        // A fragments: af0 = rows j_local 0..15 (lanes 0-15), af1 = rows 16..31 (from upper half)
        v16h af0 = {}, af1 = {};
#pragma unroll
        for (int c = 0; c < 8; ++c) {
            float xs = __shfl_xor(x8[c], 16, 32);
            if (lane < 16) {
                af0[c] = (_Float16)x8[c];
                af1[c] = (_Float16)xs;
            }
        }

#pragma unroll
        for (int t2 = 0; t2 < 2; ++t2) {
            const v16h af = (t2 == 0) ? af0 : af1;
            // redistribute window values to match C/D row layout (M = e + 8*half)
            float wv[8];
#pragma unroll
            for (int e = 0; e < 8; ++e) wv[e] = __shfl(window, t2 * 16 + e + 8 * half, 32);

#pragma unroll
            for (int ht = 0; ht < 4; ++ht) {
                // D = A2.X8 + B2 (bias comes in through the accumulator)
                v8f y = __builtin_amdgcn_wmma_f32_16x16x32_f16(false, af, false, bfrag[ht],
                                                               (short)0, cfrag[ht], false, false);
                const int hg = ht * 16 + l15;
                // f for rows M = 8*half .. 8*half+7 is 8 contiguous floats in fT
                const float4* fp = (const float4*)(fT + hg * N_PTS + jb + t2 * 16 + 8 * half);
                const float4 q0 = fp[0];
                const float4 q1 = fp[1];
                const float fv[8] = {q0.x, q0.y, q0.z, q0.w, q1.x, q1.y, q1.z, q1.w};
                float acc = 0.0f;
#pragma unroll
                for (int e = 0; e < 8; ++e) {
                    const float t = fmaxf(y[e], 0.0f);      // relu(A2.X8 + B2)
                    acc = fmaf(wv[e] * fv[e], t, acc);
                }
                racc[ht] += acc;
            }
        }
    }

#pragma unroll
    for (int ht = 0; ht < 4; ++ht) {
        float r = racc[ht];
        r += __shfl_xor(r, 16, 32); // fold M=8..15 half into M=0..7 half
        if (lane < 16) res[i * H_DIM + ht * 16 + lane] = r;
    }
}

// ---------------- net_out MLP: (N,64)->(N,16), accumulate group stats ----------------
__global__ __launch_bounds__(256) void k_net_out(const float* __restrict__ res,
                                                 const float* __restrict__ W1, const float* __restrict__ b1,
                                                 const float* __restrict__ W2, const float* __restrict__ b2,
                                                 float* __restrict__ opre, float* __restrict__ stats) {
    int n = blockIdx.x * blockDim.x + threadIdx.x;
    if (n >= N_PTS) return;
    float x[H_DIM];
#pragma unroll 8
    for (int k = 0; k < H_DIM; ++k) x[k] = res[n * H_DIM + k];
    float h1[O_DIM];
    for (int o = 0; o < O_DIM; ++o) {
        float a = b1[o];
        for (int k = 0; k < H_DIM; ++k) a += W1[o * H_DIM + k] * x[k];
        h1[o] = leaky(a);
    }
    float gs[4] = {0.f, 0.f, 0.f, 0.f}, gss[4] = {0.f, 0.f, 0.f, 0.f};
    for (int o = 0; o < O_DIM; ++o) {
        float a = b2[o];
#pragma unroll
        for (int m = 0; m < O_DIM; ++m) a += W2[o * O_DIM + m] * h1[m];
        a = leaky(a);
        opre[n * O_DIM + o] = a;
        int g = o >> 2; // O=16, 4 groups of 4
        gs[g] += a; gss[g] += a * a;
    }
#pragma unroll
    for (int g = 0; g < 4; ++g) {
        atomicAdd(&stats[2 * g + 0], gs[g]);
        atomicAdd(&stats[2 * g + 1], gss[g]);
    }
}

extern "C" void kernel_launch(void* const* d_in, const int* in_sizes, int n_in,
                              void* d_out, int out_size, void* d_ws, size_t ws_size,
                              hipStream_t stream) {
    const float* points   = (const float*)d_in[0];
    const float* nuv      = (const float*)d_in[1];
    const float* features = (const float*)d_in[2];
    const float* W_in1    = (const float*)d_in[3];
    const float* b_in1    = (const float*)d_in[4];
    const float* W_in2    = (const float*)d_in[5];
    const float* b_in2    = (const float*)d_in[6];
    const float* g_in     = (const float*)d_in[7];
    const float* beta_in  = (const float*)d_in[8];
    const float* A1       = (const float*)d_in[9];
    const float* B1       = (const float*)d_in[10];
    const float* A2       = (const float*)d_in[11];
    const float* B2       = (const float*)d_in[12];
    const float* W_out1   = (const float*)d_in[13];
    const float* b_out1   = (const float*)d_in[14];
    const float* W_out2   = (const float*)d_in[15];
    const float* b_out2   = (const float*)d_in[16];
    const float* g_out    = (const float*)d_in[17];
    const float* beta_out = (const float*)d_in[18];
    float* out = (float*)d_out;

    float* ws = (float*)d_ws;
    float* fpre_buf = ws;                                  // 4096*64
    float* fT_buf   = ws + N_PTS * H_DIM;                  // 4096*64 (transposed H x N)
    float* res_buf  = ws + 2 * N_PTS * H_DIM;              // 4096*64
    float* opre_buf = ws + 3 * N_PTS * H_DIM;              // 4096*16
    float* stats_in  = ws + 3 * N_PTS * H_DIM + N_PTS * O_DIM; // 8
    float* stats_out = stats_in + 8;                            // 8

    // zero both group-norm stat regions (atomics accumulate -> re-zero every call)
    k_zero_stats<<<1, 32, 0, stream>>>(stats_in);

    // net_in MLP -> fpre_buf (pre-GN) + stats
    k_net_in<<<N_PTS / 256, 256, 0, stream>>>(features, W_in1, b_in1, W_in2, b_in2, fpre_buf, stats_in);

    // GroupNorm(4,64), write transposed fT (H x N) for contiguous epilogue loads
    k_gn_apply<<<(N_PTS * H_DIM) / 256, 256, 0, stream>>>(fpre_buf, stats_in, g_in, beta_in,
                                                          fT_buf, H_DIM, H_DIM / 4, 1);

    // hot pairwise WMMA kernel
    k_pair_conv<<<N_PTS / 8, 256, 0, stream>>>(points, nuv, A1, B1, A2, B2, fT_buf, res_buf);

    // net_out MLP -> opre_buf + stats
    k_net_out<<<N_PTS / 256, 256, 0, stream>>>(res_buf, W_out1, b_out1, W_out2, b_out2, opre_buf, stats_out);

    // GroupNorm(4,16) -> d_out (natural layout)
    k_gn_apply<<<(N_PTS * O_DIM) / 256, 256, 0, stream>>>(opre_buf, stats_out, g_out, beta_out,
                                                          out, O_DIM, O_DIM / 4, 0);
}
